// E5CellOriginal_50783693308450
// MI455X (gfx1250) — matrix operand order repaired and checked by reading, not compiled, not run on hardware
//
#include <hip/hip_runtime.h>
#include <cmath>

// E5 cell: h_t = tanh(xp_t + (h U_h) V_h), out = h * silu(x U_z V_z)
// T=2048 B=16 D=1024 R=64, all f32 in memory, bf16 WMMA internally.

#define TT 2048
#define BB 16
#define DD 1024
#define RR 64

typedef __attribute__((ext_vector_type(16))) __bf16 v16bf;
typedef __attribute__((ext_vector_type(8)))  float  v8f;

__device__ __forceinline__ v8f v8f_zero() {
  v8f z;
#pragma unroll
  for (int j = 0; j < 8; ++j) z[j] = 0.0f;
  return z;
}

__device__ __forceinline__ v8f wmma_bf16(v16bf a, v16bf b, v8f c) {
  // D = A(16x32 bf16) * B(32x16 bf16) + C(16x16 f32)
  return __builtin_amdgcn_wmma_f32_16x16x32_bf16(
      /*neg_a=*/false, a, /*neg_b=*/false, b,
      /*c_mod=*/(short)0, c, /*reuse_a=*/false, /*reuse_b=*/false);
}

// ---- A fragment (16x32, 16-bit), ISA layout ----
// lane m = lane&15; kb = k0 + 8*(lane>=16)
// element i<8 -> K = kb+i ; element i>=8 -> K = kb+16+(i-8)
// Rows contiguous in K -> two 32B loads per lane (b128-friendly); lanes m and
// m+16 together cover 128B contiguous per row -> fully line-coalesced.
__device__ __forceinline__ v16bf loadA_f32(const float* p, int ld, int m0,
                                           int k0, int lane) {
  const int m  = m0 + (lane & 15);
  const int kb = k0 + ((lane >> 4) << 3);
  const float* r = p + (size_t)m * ld + kb;
  v16bf a;
#pragma unroll
  for (int i = 0; i < 8; ++i) a[i] = (__bf16)r[i];
#pragma unroll
  for (int i = 0; i < 8; ++i) a[8 + i] = (__bf16)r[16 + i];
  return a;
}

__device__ __forceinline__ v16bf loadA_bf16(const __bf16* p, int ld, int k0,
                                            int lane) {
  const int m  = lane & 15;
  const int kb = k0 + ((lane >> 4) << 3);
  const __bf16* r = p + (size_t)m * ld + kb;
  v16bf a;
#pragma unroll
  for (int i = 0; i < 8; ++i) a[i] = r[i];
#pragma unroll
  for (int i = 0; i < 8; ++i) a[8 + i] = r[16 + i];
  return a;
}

// ---- Swizzled B-fragment storage ----
// B fragment (32x16, 16-bit) ISA layout: lane n = lane&15,
// element i<8 -> K = k0 + 8*(lane>=16) + i ; i>=8 -> K = k0 + 16 + 8*(lane>=16) + (i-8).
// Weights stored in LDS fragment-ready: [ktile][ntile][lane][16 bf16], so a
// fragment load is one contiguous 32B read per lane (2x ds_load_b128, zero
// packing VALU).
struct Bf16x16Raw { uint4 lo, hi; };  // 32 bytes

__device__ __forceinline__ v16bf loadB_sw(const __bf16* p, int ntiles, int kt,
                                          int nt, int lane) {
  const __bf16* fp = p + (((size_t)(kt * ntiles + nt) * 32 + lane) << 4);
  Bf16x16Raw r = *(const Bf16x16Raw*)fp;
  return __builtin_bit_cast(v16bf, r);
}

// Prologue: scatter row-major f32 weights [K][N] into swizzled bf16 LDS.
// Source-major iteration keeps the global reads coalesced.
__device__ __forceinline__ void fill_B_swizzled(const float* __restrict__ src,
                                                __bf16* dst, int K, int N,
                                                int tid, int nthreads) {
  const int ntiles = N >> 4;
  for (int s = tid; s < K * N; s += nthreads) {
    const int k = s / N;  // N is a power of two -> shift
    const int n = s - k * N;
    const int kt   = k >> 5;
    const int kr   = k & 31;
    const int lane = (((kr >> 3) & 1) << 4) | (n & 15);
    const int i    = (kr & 7) | ((kr >> 4) << 3);
    const int nt   = n >> 4;
    dst[(((size_t)(kt * ntiles + nt) * 32 + lane) << 4) + i] = (__bf16)src[s];
  }
}

// -------------------------------------------------------------------------
// Fused two-stage low-rank projection over M = T*B rows:
//   GATE=false:  Out[m,:]  = (X[m,:] @ U) @ V + bias      (writes xp)
//   GATE=true :  Out[m,:] *= silu((X[m,:] @ U) @ V)       (applies z-gate)
// U (DxR) and V (RxD) staged once per block in swizzled bf16 LDS.
// -------------------------------------------------------------------------
template <bool GATE>
__global__ __launch_bounds__(256) void e5_lowrank_proj_kernel(
    const float* __restrict__ X, const float* __restrict__ U,
    const float* __restrict__ V, const float* __restrict__ bias,
    float* __restrict__ Out) {
  extern __shared__ char smem[];
  __bf16* sU = (__bf16*)smem;            // swizzled [32kt][4nt][32][16] 128 KB
  __bf16* sV = sU + DD * RR;             // swizzled [2kt][64nt][32][16] 128 KB
  float*  sA = (float*)(sV + RR * DD);   // 8 waves x [16][R]  32 KB

  const int tid  = threadIdx.x;
  const int lane = tid & 31;
  const int w    = tid >> 5;

  fill_B_swizzled(U, sU, DD, RR, tid, 256);
  fill_B_swizzled(V, sV, RR, DD, tid, 256);
  __syncthreads();

  const int n_ = lane & 15;
  const int mb = (lane >> 4) << 3;
  float* myA = sA + w * 16 * RR;

  const int mtiles = (TT * BB) / 16;  // 2048
  for (int mt = blockIdx.x * 8 + w; mt < mtiles; mt += gridDim.x * 8) {
    const int m0 = mt * 16;

    // ---- Stage A: a[16x64] = X_tile @ U  (K = D) ----
    v8f acc[4];
#pragma unroll
    for (int nt = 0; nt < 4; ++nt) acc[nt] = v8f_zero();
    for (int ks = 0; ks < DD / 32; ++ks) {
      v16bf a = loadA_f32(X, DD, m0, ks * 32, lane);
#pragma unroll
      for (int nt = 0; nt < 4; ++nt) {
        v16bf b = loadB_sw(sU, RR / 16, ks, nt, lane);
        acc[nt] = wmma_bf16(a, b, acc[nt]);
      }
    }
    // spill rank-space tile to per-wave LDS scratch (same-wave LDS in-order)
#pragma unroll
    for (int nt = 0; nt < 4; ++nt)
#pragma unroll
      for (int j = 0; j < 8; ++j)
        myA[(mb + j) * RR + nt * 16 + n_] = acc[nt][j];

    // ---- Stage B: out_tile = a @ V  (K = R = 64) ----
    v16bf a0 = loadA_f32(myA, RR, 0, 0, lane);
    v16bf a1 = loadA_f32(myA, RR, 0, 32, lane);
#pragma unroll 4
    for (int nt = 0; nt < DD / 16; ++nt) {
      const int n0 = nt * 16;
      v16bf b0 = loadB_sw(sV, DD / 16, 0, nt, lane);
      v16bf b1 = loadB_sw(sV, DD / 16, 1, nt, lane);
      v8f c = v8f_zero();
      c = wmma_bf16(a0, b0, c);
      c = wmma_bf16(a1, b1, c);
      float* orow = Out + (size_t)m0 * DD + n0 + n_;
      if (GATE) {
#pragma unroll
        for (int j = 0; j < 8; ++j) {
          const float zv = c[j];
          const float s  = zv / (1.0f + __expf(-zv));  // silu(z)
          float* pp = orow + (size_t)(mb + j) * DD;
          *pp = *pp * s;
        }
      } else {
        const float bv = bias[n0 + n_];
#pragma unroll
        for (int j = 0; j < 8; ++j)
          orow[(size_t)(mb + j) * DD] = c[j] + bv;
      }
    }
  }
}

// -------------------------------------------------------------------------
// Sequential scan: one persistent workgroup (32 waves) on one WGP.
// U_h, V_h (swizzled bf16) and h (bf16) live in LDS for all 2048 steps
// (128+128+32+4 KB of the WGP's 320 KB); the only HBM traffic on the
// critical path is the 64 KB xp_t tile per step, prefetched 2 steps ahead
// into the near cache.
// -------------------------------------------------------------------------
__global__ __launch_bounds__(1024) void e5_scan_kernel(
    const float* __restrict__ Uh, const float* __restrict__ Vh,
    const float* __restrict__ h0, float* __restrict__ HXP,
    float* __restrict__ hlast) {
  extern __shared__ char smem[];
  __bf16* sU   = (__bf16*)smem;              // swizzled [D][R]  128 KB
  __bf16* sV   = sU + DD * RR;               // swizzled [R][D]  128 KB
  __bf16* sH   = sV + RR * DD;               // [B][D] row-major  32 KB
  float*  sTmp = (float*)(sH + BB * DD);     // [B][R]             4 KB

  const int tid  = threadIdx.x;
  const int lane = tid & 31;
  const int w    = tid >> 5;

  fill_B_swizzled(Uh, sU, DD, RR, tid, 1024);
  fill_B_swizzled(Vh, sV, RR, DD, tid, 1024);
  for (int i = tid; i < BB * DD; i += 1024) sH[i] = (__bf16)h0[i];
  __syncthreads();

  const int n_ = lane & 15;
  const int mb = (lane >> 4) << 3;

  for (int t = 0; t < TT; ++t) {
    float* xpt = HXP + (size_t)t * BB * DD;
    // Prefetch xp two steps ahead into the NEAR cache (locality 3 -> WGP
    // scope): 64 KB = 1024 threads x one 64B line each; ~128 KB in flight.
    if (t + 2 < TT)
      __builtin_prefetch((const char*)(xpt + 2 * BB * DD) + (size_t)tid * 64,
                         0, 3);

    sTmp[tid] = 0.0f;  // B*R == 1024 == blockDim
    __syncthreads();

    // ---- Stage A: tmp = h @ U_h; wave w owns K-chunk [32w, 32w+32) ----
    {
      v16bf a = loadA_bf16(sH, DD, w * 32, lane);
#pragma unroll
      for (int nt = 0; nt < 4; ++nt) {
        v16bf b = loadB_sw(sU, RR / 16, w, nt, lane);
        v8f c = v8f_zero();
        c = wmma_bf16(a, b, c);
#pragma unroll
        for (int j = 0; j < 8; ++j)
          atomicAdd(&sTmp[(mb + j) * RR + nt * 16 + n_], c[j]);  // ds_add_f32
      }
    }
    __syncthreads();

    // ---- Stage B: h_new = tanh(xp_t + tmp @ V_h); wave owns 2 N-tiles ----
    v16bf a0 = loadA_f32(sTmp, RR, 0, 0, lane);
    v16bf a1 = loadA_f32(sTmp, RR, 0, 32, lane);
#pragma unroll
    for (int u = 0; u < 2; ++u) {
      const int ntile = w * 2 + u;
      const int n0 = ntile * 16;
      v16bf b0 = loadB_sw(sV, DD / 16, 0, ntile, lane);
      v16bf b1 = loadB_sw(sV, DD / 16, 1, ntile, lane);
      v8f c = v8f_zero();
      c = wmma_bf16(a0, b0, c);
      c = wmma_bf16(a1, b1, c);
#pragma unroll
      for (int j = 0; j < 8; ++j) {
        const int m = mb + j;
        const size_t idx = (size_t)m * DD + n0 + n_;  // valid in [B][D] tile
        const float hn = tanhf(xpt[idx] + c[j]);
        xpt[idx] = hn;             // overwrite xp_t with h_t (lane-exclusive)
        sH[idx]  = (__bf16)hn;     // bf16 state for next step's A operand
        if (t == TT - 1) hlast[idx] = hn;
      }
    }
    __syncthreads();
  }
}

extern "C" void kernel_launch(void* const* d_in, const int* in_sizes, int n_in,
                              void* d_out, int out_size, void* d_ws,
                              size_t ws_size, hipStream_t stream) {
  (void)in_sizes; (void)n_in; (void)d_ws; (void)ws_size; (void)out_size;
  const float* x    = (const float*)d_in[0];
  const float* h0   = (const float*)d_in[1];
  const float* Uh   = (const float*)d_in[2];
  const float* Vh   = (const float*)d_in[3];
  const float* Ux   = (const float*)d_in[4];
  const float* Vx   = (const float*)d_in[5];
  const float* Uz   = (const float*)d_in[6];
  const float* Vz   = (const float*)d_in[7];
  const float* bias = (const float*)d_in[8];

  float* out   = (float*)d_out;                  // [T,B,D] then [B,D]
  float* hlast = out + (size_t)TT * BB * DD;

  const size_t ldsProj =
      (size_t)(DD * RR + RR * DD) * 2 + (size_t)8 * 16 * RR * sizeof(float);
  const size_t ldsScan =
      (size_t)(DD * RR + RR * DD + BB * DD) * 2 + (size_t)BB * RR * sizeof(float);

  // 1) xp = x @ U_x @ V_x + b  -> stored in d_out (consumed & replaced by scan)
  e5_lowrank_proj_kernel<false><<<128, 256, ldsProj, stream>>>(x, Ux, Vx, bias, out);
  // 2) sequential recurrence, fully LDS-resident; out <- h_t in place
  e5_scan_kernel<<<1, 1024, ldsScan, stream>>>(Uh, Vh, h0, out, hlast);
  // 3) z = x @ U_z @ V_z ; out *= silu(z)
  e5_lowrank_proj_kernel<true><<<128, 256, ldsProj, stream>>>(x, Uz, Vz, bias, out);
}
